// PairwiseScore_90615220011455
// MI455X (gfx1250) — matrix operand order repaired and checked by reading, not compiled, not run on hardware
//
#include <hip/hip_runtime.h>
#include <hip/hip_bf16.h>

typedef __attribute__((ext_vector_type(16))) __bf16 v16bf;
typedef __attribute__((ext_vector_type(8)))  float  v8f;

#define EE    256
#define HID   150
#define HIDP  160
#define NB    2
#define NN    384
#define ROWS  (NB * NN)   // 768
#define NT    10          // HIDP / 16 column tiles
#define KT1   8           // EE / 32 k-steps for GEMM1
#define KT2   5           // HIDP / 32 k-steps for GEMM2

// ------- prep: P[row,h] = x_row @ W1[0:256,h] + b1[h]   (padded to HIDP, pad = 0)
//               Qt[h,row] = x_row @ W1[256:512,h]        (transposed for vector loads)
__global__ void prep_pq(const float* __restrict__ x, const float* __restrict__ W1,
                        const float* __restrict__ b1,
                        float* __restrict__ P, float* __restrict__ Qt) {
    int id = blockIdx.x * blockDim.x + threadIdx.x;
    const int total = 2 * ROWS * HIDP;
    if (id >= total) return;
    int part = id / (ROWS * HIDP);
    int rem  = id % (ROWS * HIDP);
    int row  = rem / HIDP;
    int h    = rem % HIDP;
    float acc = 0.0f;
    if (h < HID) {
        const float* xr = x + (size_t)row * EE;
        const float* w  = W1 + (size_t)(part * EE) * HID + h;
        for (int e = 0; e < EE; ++e) acc += xr[e] * w[(size_t)e * HID];
        if (part == 0) acc += b1[h];
    }
    if (part == 0) P[(size_t)row * HIDP + h] = acc;
    else           Qt[(size_t)h * ROWS + row] = acc;
}

// ------- prep: swizzle W1c / W2 into per-lane WMMA B-fragment order --------------------
// B fragment (16-bit, 32x16): lane L holds N = L%16; element t (0..15) holds K = 8*(L/16) + (t&7) + 16*(t>>3).
// Fragment storage: [n_tile][k_step][lane][t] contiguous -> one 32B v16bf load per lane.
__global__ void prep_w(const float* __restrict__ W1, const float* __restrict__ W2,
                       const float* __restrict__ b2, const float* __restrict__ W3,
                       __bf16* __restrict__ W1s, __bf16* __restrict__ W2s,
                       float* __restrict__ b2p, float* __restrict__ W3p) {
    int id = blockIdx.x * blockDim.x + threadIdx.x;
    const int n1 = NT * KT1 * 32 * 16;           // 40960
    const int n2 = NT * KT2 * 32 * 16;           // 25600
    if (id < n1) {
        int t = id & 15, lane = (id >> 4) & 31, f = id >> 9;
        int kt = f & 7, nt = f >> 3;
        int k = 32 * kt + 8 * (lane >> 4) + (t & 7) + 16 * (t >> 3);  // 0..255  (rows of W1c)
        int n = nt * 16 + (lane & 15);           // 0..159
        float v = (n < HID) ? W1[(size_t)(512 + k) * HID + n] : 0.0f;
        W1s[id] = (__bf16)v;
    } else if (id < n1 + n2) {
        int j = id - n1;
        int t = j & 15, lane = (j >> 4) & 31, f = j >> 9;
        int kt = f % 5, nt = f / 5;
        int k = 32 * kt + 8 * (lane >> 4) + (t & 7) + 16 * (t >> 3);  // 0..159
        int n = nt * 16 + (lane & 15);
        float v = (k < HID && n < HID) ? W2[(size_t)k * HID + n] : 0.0f;
        W2s[j] = (__bf16)v;
    } else {
        int j = id - (n1 + n2);
        if (j < HIDP)            b2p[j] = (j < HID) ? b2[j] : 0.0f;
        else if (j < 2 * HIDP) { int t = j - HIDP; W3p[t] = (t < HID) ? W3[t] : 0.0f; }
    }
}

// ------- main: per block = (b, i, 64 j's); 4 waves x 16 j's ----------------------------
__launch_bounds__(128)
__global__ void pair_main(const float* __restrict__ x, const float* __restrict__ m,
                          const __bf16* __restrict__ W1s, const __bf16* __restrict__ W2s,
                          const float* __restrict__ P, const float* __restrict__ Qt,
                          const float* __restrict__ b2p, const float* __restrict__ W3p,
                          const float* __restrict__ b3, float* __restrict__ out) {
    const int b    = blockIdx.z;
    const int i    = blockIdx.y;
    const int jb   = blockIdx.x;
    const int tid  = threadIdx.x;
    const int wave = tid >> 5;
    const int lane = tid & 31;
    const int half = lane >> 4;    // 0: lanes 0-15, 1: lanes 16-31
    const int lm   = lane & 15;
    const int j0   = jb * 64 + wave * 16;

    __shared__ float  xi_s[EE];
    __shared__ __bf16 h1_s[4][16][HIDP];

    const float* xi = x + ((size_t)b * NN + i) * EE;
    for (int e = tid; e < EE; e += 128) xi_s[e] = xi[e];
    __syncthreads();

    // ---- A fragments for GEMM1: A[j, e] = bf16(x_j[e] * x_i[e]) ----
    // 16-bit A 16x32 layout: lane holds M = lane%16; t=0..7 -> K = 32kt + 8*half + t,
    //                                              t=8..15 -> K = 32kt + 8*half + 16 + (t-8)
    const int jj = j0 + lm;
    const float* xj = x + ((size_t)b * NN + jj) * EE;
    v16bf a1[KT1];
#pragma unroll
    for (int kt = 0; kt < KT1; ++kt) {
        int b1i = 32 * kt + 8 * half;
#pragma unroll
        for (int t = 0; t < 8; ++t) {
            a1[kt][t]     = (__bf16)(xj[b1i + t]      * xi_s[b1i + t]);
            a1[kt][t + 8] = (__bf16)(xj[b1i + 16 + t] * xi_s[b1i + 16 + t]);
        }
    }

    // ---- GEMM1: h1 = relu(P_i(+b1) + Q_j + cross) ----
    const size_t rowP = ((size_t)b * NN + i) * HIDP;
    const int    jrow = b * NN + j0 + 8 * half;          // first of 8 consecutive j rows
    for (int nt = 0; nt < NT; ++nt) {
        int h = nt * 16 + lm;                            // C/D layout: N = lane%16
        float base = P[rowP + h];                        // includes b1
        const float* q = Qt + (size_t)h * ROWS + jrow;   // 8 consecutive floats
        float4 qa = *(const float4*)(q);
        float4 qb = *(const float4*)(q + 4);
        v8f acc;
        acc[0] = base + qa.x; acc[1] = base + qa.y;
        acc[2] = base + qa.z; acc[3] = base + qa.w;
        acc[4] = base + qb.x; acc[5] = base + qb.y;
        acc[6] = base + qb.z; acc[7] = base + qb.w;
#pragma unroll
        for (int kt = 0; kt < KT1; ++kt) {
            v16bf bf = *(const v16bf*)(W1s + ((size_t)(nt * KT1 + kt) * 32 + lane) * 16);
            acc = __builtin_amdgcn_wmma_f32_16x16x32_bf16(false, a1[kt], false, bf,
                                                          (short)0, acc, false, false);
        }
#pragma unroll
        for (int r = 0; r < 8; ++r) {                    // C/D layout: M = r + 8*half
            float v = acc[r] > 0.0f ? acc[r] : 0.0f;
            h1_s[wave][r + 8 * half][h] = (__bf16)v;
        }
    }
    __syncthreads();

    // ---- A fragments for GEMM2 from LDS h1 ----
    v16bf a2[KT2];
#pragma unroll
    for (int kt = 0; kt < KT2; ++kt) {
        int b1i = 32 * kt + 8 * half;
#pragma unroll
        for (int t = 0; t < 8; ++t) {
            a2[kt][t]     = h1_s[wave][lm][b1i + t];
            a2[kt][t + 8] = h1_s[wave][lm][b1i + 16 + t];
        }
    }

    // ---- GEMM2 + fold in W3: spart[r] = sum_h relu(h2[j,h]) * W3[h] ----
    float spart[8];
#pragma unroll
    for (int r = 0; r < 8; ++r) spart[r] = 0.0f;

    for (int nt = 0; nt < NT; ++nt) {
        int h = nt * 16 + lm;
        float b2v = b2p[h];
        v8f acc;
#pragma unroll
        for (int r = 0; r < 8; ++r) acc[r] = b2v;
#pragma unroll
        for (int kt = 0; kt < KT2; ++kt) {
            v16bf bf = *(const v16bf*)(W2s + ((size_t)(nt * KT2 + kt) * 32 + lane) * 16);
            acc = __builtin_amdgcn_wmma_f32_16x16x32_bf16(false, a2[kt], false, bf,
                                                          (short)0, acc, false, false);
        }
        float w3 = W3p[h];
#pragma unroll
        for (int r = 0; r < 8; ++r) {
            float v = acc[r] > 0.0f ? acc[r] : 0.0f;
            spart[r] += v * w3;
        }
    }

    // reduce over the 16 h-lanes within each half (xor of bits 0..3 stays in-half)
#pragma unroll
    for (int r = 0; r < 8; ++r) {
        float v = spart[r];
        v += __shfl_xor(v, 1, 32);
        v += __shfl_xor(v, 2, 32);
        v += __shfl_xor(v, 4, 32);
        v += __shfl_xor(v, 8, 32);
        spart[r] = v;
    }

    if (lm == 0) {
        float b3v = b3[0];
        float mi  = m[(size_t)b * NN + i];
        size_t obase = ((size_t)b * NN + i) * NN + j0 + 8 * half;   // 8 consecutive j's
        float res[8];
#pragma unroll
        for (int r = 0; r < 8; ++r) {
            int j = j0 + r + 8 * half;
            res[r] = (mi + m[(size_t)b * NN + j] + spart[r] + b3v) * (1.0f / 3.0f);
        }
        *(float4*)(out + obase)     = make_float4(res[0], res[1], res[2], res[3]);
        *(float4*)(out + obase + 4) = make_float4(res[4], res[5], res[6], res[7]);
    }
}

// ---------------------------------------------------------------------------------------
extern "C" void kernel_launch(void* const* d_in, const int* in_sizes, int n_in,
                              void* d_out, int out_size, void* d_ws, size_t ws_size,
                              hipStream_t stream) {
    const float* x  = (const float*)d_in[0];   // (B,N,E)
    const float* ms = (const float*)d_in[1];   // (B,N,1)
    const float* W1 = (const float*)d_in[2];   // (768,150)
    const float* b1 = (const float*)d_in[3];   // (150,)
    const float* W2 = (const float*)d_in[4];   // (150,150)
    const float* b2 = (const float*)d_in[5];   // (150,)
    const float* W3 = (const float*)d_in[6];   // (150,1)
    const float* b3 = (const float*)d_in[7];   // (1,)
    float* out = (float*)d_out;

    char* ws = (char*)d_ws;
    size_t off = 0;
    auto alloc = [&](size_t bytes) { char* p = ws + off; off = (off + bytes + 255) & ~(size_t)255; return p; };
    __bf16* W1s = (__bf16*)alloc((size_t)NT * KT1 * 32 * 16 * sizeof(__bf16)); // 80 KB
    __bf16* W2s = (__bf16*)alloc((size_t)NT * KT2 * 32 * 16 * sizeof(__bf16)); // 50 KB
    float*  P   = (float*) alloc((size_t)ROWS * HIDP * sizeof(float));         // 480 KB
    float*  Qt  = (float*) alloc((size_t)HIDP * ROWS * sizeof(float));         // 480 KB
    float*  b2p = (float*) alloc(HIDP * sizeof(float));
    float*  W3p = (float*) alloc(HIDP * sizeof(float));
    (void)ws_size; (void)in_sizes; (void)n_in; (void)out_size;

    {   // P (with b1 folded) and transposed Q precompute
        int total = 2 * ROWS * HIDP;
        prep_pq<<<(total + 255) / 256, 256, 0, stream>>>(x, W1, b1, P, Qt);
    }
    {   // weight swizzle + padded b2 / W3 vectors
        int total = NT * KT1 * 32 * 16 + NT * KT2 * 32 * 16 + 2 * HIDP;
        prep_w<<<(total + 255) / 256, 256, 0, stream>>>(W1, W2, b2, W3, W1s, W2s, b2p, W3p);
    }
    {   // main pairwise kernel: grid (j-tiles of 64, i, b), 128 threads = 4 waves
        dim3 grid(NN / 64, NN, NB);
        pair_main<<<grid, 128, 0, stream>>>(x, ms, W1s, W2s, P, Qt, b2p, W3p, b3, out);
    }
}
